// UltraLSNTLiteForecaster_87875030876706
// MI455X (gfx1250) — compile-verified, matrix operand order checked by reading, not run on hardware
//
#include <hip/hip_runtime.h>

// ---------------------------------------------------------------------------
// UltraLSNTLiteForecaster forward for MI455X (gfx1250, wave32, WMMA bf16,
// async global->LDS staging when available)
// ---------------------------------------------------------------------------

typedef __bf16 bf16;
typedef __attribute__((ext_vector_type(16))) __bf16 v16bf;
typedef __attribute__((ext_vector_type(8)))  float  v8f;
typedef int async_v4i __attribute__((vector_size(16)));

#define DEV static __device__ __forceinline__

// Model constants
constexpr int B_    = 256;
constexpr int S_    = 512;
constexpr int D_    = 32;
constexpr int H_    = 256;
constexpr int E_    = 8;
constexpr int NB_   = 2;
constexpr int PRED_ = 96;
constexpr int NH_   = 4;
constexpr int HD_   = 64;      // H/NH
constexpr int BS_   = B_ * S_; // 131072
constexpr int H3_   = 3 * H_;  // 768

DEV float gelu_exact(float x) {
  return 0.5f * x * (1.0f + erff(x * 0.70710678118654752440f));
}

// ---------------------------------------------------------------------------
// CDNA5 async global->LDS copy (GLOBAL_LOAD_ASYNC_TO_LDS_B128, ASYNCcnt).
// Guarded: falls back to plain 16B copies if the builtin isn't declared.
// Probe-learned signature: (as1 v4i* gsrc, as3 v4i* ldst, imm off, imm cpol)
// ---------------------------------------------------------------------------
#if defined(__AMDGCN__) && __has_builtin(__builtin_amdgcn_global_load_async_to_lds_b128) && __has_builtin(__builtin_amdgcn_s_wait_asynccnt)
#define ASYNC_LDS 1
#else
#define ASYNC_LDS 0
#endif

// copy 8 bf16 (16 bytes), 16B-aligned on both sides
DEV void lds_copy8(bf16* ldst, const bf16* gsrc) {
#if ASYNC_LDS
  __builtin_amdgcn_global_load_async_to_lds_b128(
      (__attribute__((address_space(1))) async_v4i*)(const void*)gsrc,
      (__attribute__((address_space(3))) async_v4i*)(void*)ldst,
      /*offset=*/0, /*cpol=*/0);
#else
  struct U16B { unsigned long long x, y; };
  *(U16B*)ldst = *(const U16B*)gsrc;
#endif
}

DEV void async_wait0() {
#if ASYNC_LDS
  __builtin_amdgcn_s_wait_asynccnt(0);
#endif
}

// ---------------------------------------------------------------------------
// WMMA helpers (CDNA5 16x16x32 bf16, f32 accumulate)
// Fragment layout per ISA 7.12.2 (16-bit A 16x32):
//   lane l: row m = l&15, half hi = l>>4 (k offset +8)
//   VGPR g<4 : packed pair k = 2g,2g+1 ; g>=4 : k = 16+2(g-4), +1
// B fragment mirrors A with n = l&15, so B tiles are kept transposed ([n][k]).
// ---------------------------------------------------------------------------
DEV v16bf frag_load_pairs(const bf16* tile, int ld, int lane) {
  const int m  = lane & 15;
  const int hi = lane >> 4;
  const bf16* rp = tile + m * ld + hi * 8;
  v16bf f;
#pragma unroll
  for (int g = 0; g < 8; ++g) {
    const int kb = (g < 4) ? (2 * g) : (16 + 2 * (g - 4));
    f[2 * g]     = rp[kb];
    f[2 * g + 1] = rp[kb + 1];
  }
  return f;
}

DEV v8f wmma_bf16(v16bf a, v16bf b, v8f c) {
  return __builtin_amdgcn_wmma_f32_16x16x32_bf16(
      /*neg_a=*/false, a, /*neg_b=*/false, b,
      /*c_mod=*/(short)0, c, /*reuse_a=*/false, /*reuse_b=*/false);
}

DEV v8f zero8() {
  v8f z;
#pragma unroll
  for (int i = 0; i < 8; ++i) z[i] = 0.0f;
  return z;
}

// ---------------------------------------------------------------------------
// Generic tiled WMMA GEMM, 64x64 block tile, 128 threads (4 waves stacked in
// M; each wave computes 16x64 = 4 accumulators -> 4 WMMA per K-step).
// B ("Bt") is stored transposed [N][K] so both A and B staging are contiguous
// 16B rows -> async global->LDS.
//   AMODE 0: A is bf16 row-major [M][K] (lda == K)
//   AMODE 1: A is conv im2col over xsrc[(b*512 + t)*256 + i], K = k1*256,
//            tap = kchunk>>8, src row t' = t + tap - conv_pad (zero padded)
//   OUTMODE 0: f32 store   1: bf16 store
//   OUTMODE 2: o-proj epilogue: v += resid(bf16); atomic mean into accum(B,H)
// Batched over blockIdx.z via strideA/strideB/strideC/strideBias (elements).
// ---------------------------------------------------------------------------
template <int AMODE, int OUTMODE, bool BIAS, bool GELU>
__global__ void gemm_wmma_bf16(const bf16* __restrict__ A, int lda,
                               const bf16* __restrict__ Bt, int ldb, // == K
                               const float* __restrict__ bias,
                               void* __restrict__ Cout, int ldc,
                               int M, int N, int K, int conv_pad,
                               const bf16* __restrict__ resid,
                               float* __restrict__ accum,
                               long strideA, long strideB, long strideC,
                               int strideBias) {
  constexpr int BM = 64, BN = 64, BK = 32;
  __shared__ alignas(16) bf16 As[BM][BK];
  __shared__ alignas(16) bf16 Bts[BN][BK];

  const int tid  = threadIdx.x;      // 0..127
  const int wave = tid >> 5;         // 0..3 -> M strip
  const int lane = tid & 31;
  const int m0   = blockIdx.y * BM;
  const int n0   = blockIdx.x * BN;
  const int e    = blockIdx.z;

  const bf16*  Ap    = A  + (long)e * strideA;
  const bf16*  Bp    = Bt + (long)e * strideB;
  const float* biasp = BIAS ? (bias + (long)e * strideBias) : nullptr;

  const int rstage = tid >> 1;        // 0..63
  const int cstage = (tid & 1) * 16;  // 0 or 16

  v8f acc[4];
#pragma unroll
  for (int f = 0; f < 4; ++f) acc[f] = zero8();

  for (int k0 = 0; k0 < K; k0 += BK) {
    // ---- stage A tile (64x32): 16 contiguous bf16 per thread ----
    if (AMODE == 0) {
      const bf16* src = Ap + (long)(m0 + rstage) * lda + k0 + cstage;
      lds_copy8(&As[rstage][cstage], src);
      lds_copy8(&As[rstage][cstage + 8], src + 8);
    } else {
      const int rg  = m0 + rstage;
      const int bb  = rg >> 9;          // /S_
      const int t   = rg & 511;
      const int tap = k0 >> 8;          // chunk never straddles taps (H=256)
      const int i0c = (k0 & 255) + cstage;
      const int tp  = t + tap - conv_pad;
      if (tp >= 0 && tp < S_) {
        const bf16* src = Ap + ((long)(bb * S_ + tp) * H_ + i0c);
        lds_copy8(&As[rstage][cstage], src);
        lds_copy8(&As[rstage][cstage + 8], src + 8);
      } else {
#pragma unroll
        for (int j = 0; j < 16; ++j) As[rstage][cstage + j] = (bf16)0.0f;
      }
    }
    // ---- stage Bt tile (64x32): Bt is [N][K], contiguous per row ----
    {
      const int gn = n0 + rstage;
      if (gn < N) {
        const bf16* src = Bp + (long)gn * ldb + k0 + cstage;
        lds_copy8(&Bts[rstage][cstage], src);
        lds_copy8(&Bts[rstage][cstage + 8], src + 8);
      } else {
#pragma unroll
        for (int j = 0; j < 16; ++j) Bts[rstage][cstage + j] = (bf16)0.0f;
      }
    }
    async_wait0();
    __syncthreads();

    const v16bf a = frag_load_pairs(&As[wave * 16][0], BK, lane);
#pragma unroll
    for (int f = 0; f < 4; ++f) {
      const v16bf b = frag_load_pairs(&Bts[f * 16][0], BK, lane);
      acc[f] = wmma_bf16(a, b, acc[f]);
    }
    __syncthreads();
  }

  // ---- epilogue ----
  const int nlo = lane & 15;
  const int hi  = lane >> 4;
#pragma unroll
  for (int f = 0; f < 4; ++f) {
    const int col = n0 + f * 16 + nlo;
    if (col >= N) continue;
    const float bv = BIAS ? biasp[col] : 0.0f;
#pragma unroll
    for (int i = 0; i < 8; ++i) {
      const int row = m0 + wave * 16 + hi * 8 + i;
      float v = acc[f][i] + bv;
      if (GELU) v = gelu_exact(v);
      if (OUTMODE == 0) {
        ((float*)Cout)[(long)e * strideC + (long)row * ldc + col] = v;
      } else if (OUTMODE == 1) {
        ((bf16*)Cout)[(long)e * strideC + (long)row * ldc + col] = (bf16)v;
      } else {
        v += (float)resid[(long)row * H_ + col];
        atomicAdd(&accum[(row >> 9) * H_ + col], v * (1.0f / (float)S_));
      }
    }
  }
}

// ---------------------------------------------------------------------------
// Attention: one workgroup per (b, head, 16-row q block). 128 threads.
// Q.K^T (WMMA, K frags read directly from head-major K buffer), bf16 scores
// in LDS, softmax with deferred 1/rowsum, probs.V via WMMA with per-chunk
// V transposition in LDS.
// ---------------------------------------------------------------------------
__global__ void attention_kernel(const bf16* __restrict__ qh,
                                 const bf16* __restrict__ kh,
                                 const bf16* __restrict__ vh,
                                 bf16* __restrict__ ao) {
  const int qb = blockIdx.x;        // 0..31
  const int bh = blockIdx.y;        // 0..1023
  const int b  = bh >> 2;
  const int h  = bh & 3;
  const int tid  = threadIdx.x;     // 0..127
  const int wave = tid >> 5;
  const int lane = tid & 31;

  __shared__ alignas(16) bf16 qs[16 * 64];
  __shared__ alignas(16) bf16 pb[16 * 512];
  __shared__ alignas(16) bf16 vts[64 * 32];
  __shared__ float red[16 * 8];
  __shared__ float rmax[16];
  __shared__ float rsum[16];

  const long headBase = (long)bh * S_ * HD_;

  // q tile -> LDS (async: exactly 128 lanes x 16B)
  lds_copy8(qs + tid * 8, qh + headBase + (long)qb * 16 * HD_ + tid * 8);
  async_wait0();
  __syncthreads();

  // scores = (q k^T) / 8 ; each wave does 8 column tiles of 16 keys
  for (int j = 0; j < 8; ++j) {
    const int ct = wave * 8 + j;
    const bf16* kt = kh + headBase + (long)ct * 16 * HD_;
    v8f acc = zero8();
    {
      const v16bf a0 = frag_load_pairs(qs, 64, lane);
      const v16bf b0 = frag_load_pairs(kt, 64, lane);
      acc = wmma_bf16(a0, b0, acc);
      const v16bf a1 = frag_load_pairs(qs + 32, 64, lane);
      const v16bf b1 = frag_load_pairs(kt + 32, 64, lane);
      acc = wmma_bf16(a1, b1, acc);
    }
    const int n = lane & 15, hi = lane >> 4;
#pragma unroll
    for (int i = 0; i < 8; ++i)
      pb[(hi * 8 + i) * 512 + ct * 16 + n] = (bf16)(acc[i] * 0.125f);
  }
  __syncthreads();

  // softmax over each of 16 rows (8 threads / row); probs left unnormalized
  {
    const int r = tid >> 3, c8 = tid & 7;
    float mx = -3.0e38f;
    for (int c = c8; c < 512; c += 8) mx = fmaxf(mx, (float)pb[r * 512 + c]);
    red[r * 8 + c8] = mx;
    __syncthreads();
    if (c8 == 0) {
      float m = red[r * 8];
      for (int q = 1; q < 8; ++q) m = fmaxf(m, red[r * 8 + q]);
      rmax[r] = m;
    }
    __syncthreads();
    const float m = rmax[r];
    float s = 0.0f;
    for (int c = c8; c < 512; c += 8) {
      const float ev = __expf((float)pb[r * 512 + c] - m);
      pb[r * 512 + c] = (bf16)ev;
      s += ev;
    }
    red[r * 8 + c8] = s;
    __syncthreads();
    if (c8 == 0) {
      float t = 0.0f;
      for (int q = 0; q < 8; ++q) t += red[r * 8 + q];
      rsum[r] = t;
    }
    __syncthreads();
  }

  // probs @ V : wave w owns head-dim columns [w*16, w*16+16)
  v8f acc = zero8();
  for (int kc = 0; kc < 16; ++kc) {
    // stage transposed v chunk: vts[d][k], keys kc*32..kc*32+31
    for (int j = tid; j < 32 * 64; j += 128) {
      const int kk = j >> 6, dd = j & 63;
      vts[dd * 32 + kk] = vh[headBase + (long)(kc * 32 + kk) * HD_ + dd];
    }
    __syncthreads();
    const v16bf a   = frag_load_pairs(pb + kc * 32, 512, lane);
    const v16bf bfr = frag_load_pairs(vts + (wave * 16) * 32, 32, lane);
    acc = wmma_bf16(a, bfr, acc);
    __syncthreads();
  }
  {
    const int n = lane & 15, hi = lane >> 4;
#pragma unroll
    for (int i = 0; i < 8; ++i) {
      const int m = hi * 8 + i;
      const float v = acc[i] / rsum[m];
      const int srow = qb * 16 + m;
      ao[((long)(b * S_ + srow)) * H_ + h * HD_ + wave * 16 + n] = (bf16)v;
    }
  }
}

// ---------------------------------------------------------------------------
// Small elementwise / reduction kernels
// ---------------------------------------------------------------------------
DEV float block_reduce_sum256(float v, float* sb) {
  const int t = threadIdx.x;
  sb[t] = v;
  __syncthreads();
  for (int s = 128; s > 0; s >>= 1) {
    if (t < s) sb[t] += sb[t + s];
    __syncthreads();
  }
  const float r = sb[0];
  __syncthreads();
  return r;
}

__global__ void cvt_f32_bf16(const float* __restrict__ src,
                             bf16* __restrict__ dst, long n) {
  const long i = (long)blockIdx.x * 256 + threadIdx.x;
  if (i < n) dst[i] = (bf16)src[i];
}

// batched: src [e][K][N] f32 -> dst [e][N][K] bf16  (Bt layout)
__global__ void transpose_bt_bf16(const float* __restrict__ src,
                                  bf16* __restrict__ dst, int K, int N,
                                  long total) {
  const long i = (long)blockIdx.x * 256 + threadIdx.x;
  if (i < total) {
    const long kn = (long)K * N;
    const long e = i / kn;
    const long r = i - e * kn;
    const int k = (int)(r / N), n = (int)(r % N);
    dst[e * kn + (long)n * K + k] = (bf16)src[i];
  }
}

// conv weight (O=256, I=256, k1) -> Bt layout dst[o][(kk*256 + ii)]
__global__ void conv_repack_bt_bf16(const float* __restrict__ src,
                                    bf16* __restrict__ dst, int k1) {
  const long total = (long)H_ * H_ * k1;
  const long i = (long)blockIdx.x * 256 + threadIdx.x;
  if (i < total) {
    const int o  = (int)(i / (H_ * k1));
    const int r  = (int)(i % (H_ * k1));
    const int ii = r / k1;
    const int kk = r % k1;
    dst[(long)o * (k1 * H_) + kk * H_ + ii] = (bf16)src[i];
  }
}

__global__ void zero_f32(float* p, long n) {
  const long i = (long)blockIdx.x * 256 + threadIdx.x;
  if (i < n) p[i] = 0.0f;
}

// LN over H=256 for bf16 rows (encoder fusion LN). 1 block / row.
__global__ void ln_bf16_kernel(const bf16* __restrict__ in,
                               const float* __restrict__ g,
                               const float* __restrict__ be,
                               bf16* __restrict__ out) {
  __shared__ float sb[256];
  const long row = blockIdx.x;
  const int  t   = threadIdx.x;
  const float x = (float)in[row * H_ + t];
  const float m = block_reduce_sum256(x, sb) * (1.0f / H_);
  const float d = x - m;
  const float v = block_reduce_sum256(d * d, sb) * (1.0f / H_);
  const float y = d * rsqrtf(v + 1e-5f) * g[t] + be[t];
  out[row * H_ + t] = (bf16)y;
}

// LN for f32 rows, dual f32+bf16 output (block ln1)
__global__ void ln1_kernel(const float* __restrict__ in,
                           const float* __restrict__ g,
                           const float* __restrict__ be,
                           float* __restrict__ outf,
                           bf16* __restrict__ outb) {
  __shared__ float sb[256];
  const int row = blockIdx.x, t = threadIdx.x;
  const float x = in[row * H_ + t];
  const float m = block_reduce_sum256(x, sb) * (1.0f / H_);
  const float d = x - m;
  const float v = block_reduce_sum256(d * d, sb) * (1.0f / H_);
  const float y = d * rsqrtf(v + 1e-5f) * g[t] + be[t];
  outf[row * H_ + t] = y;
  outb[row * H_ + t] = (bf16)y;
}

// out = LN(a + b) * g + be + res
__global__ void ln2add_kernel(const float* __restrict__ a,
                              const float* __restrict__ bsrc,
                              const float* __restrict__ g,
                              const float* __restrict__ be,
                              const float* __restrict__ res,
                              float* __restrict__ out) {
  __shared__ float sb[256];
  const int row = blockIdx.x, t = threadIdx.x;
  const float x = a[row * H_ + t] + bsrc[row * H_ + t];
  const float m = block_reduce_sum256(x, sb) * (1.0f / H_);
  const float d = x - m;
  const float v = block_reduce_sum256(d * d, sb) * (1.0f / H_);
  const float y = d * rsqrtf(v + 1e-5f) * g[t] + be[t];
  out[row * H_ + t] = y + res[row * H_ + t];
}

// scatter qkv (B*S, 768) -> head-major q/k/v (B, NH, S, HD)
__global__ void qkv_reorder(const bf16* __restrict__ qkv,
                            bf16* __restrict__ qh, bf16* __restrict__ kh,
                            bf16* __restrict__ vh) {
  const long i = (long)blockIdx.x * 256 + threadIdx.x;
  if (i >= (long)BS_ * H3_) return;
  const long r = i / H3_;
  const int  c = (int)(i % H3_);
  const int which = c >> 8;
  const int w256  = c & 255;
  const int h = w256 >> 6;
  const int d = w256 & 63;
  const int b = (int)(r >> 9);
  const int s = (int)(r & 511);
  const long dst = (((long)b * NH_ + h) * S_ + s) * HD_ + d;
  bf16* dp = (which == 0) ? qh : (which == 1) ? kh : vh;
  dp[dst] = qkv[i];
}

// MoE router: softmax -> top2 gates, usage/counts, aux loss accumulation
__global__ void moe_gate_kernel(const float* __restrict__ xnf,
                                const float* __restrict__ rtw,
                                float* __restrict__ gates,
                                float* __restrict__ auxslot) {
  __shared__ float usage[E_];
  __shared__ float counts[E_];
  const int t = threadIdx.x;  // 256 = one row per thread
  if (t < E_) { usage[t] = 0.0f; counts[t] = 0.0f; }
  __syncthreads();
  const float* xr = xnf + (long)t * H_;
  float lg[E_];
  float mx = -3.0e38f;
#pragma unroll
  for (int e = 0; e < E_; ++e) {
    float s = 0.0f;
    for (int k = 0; k < H_; ++k) s += xr[k] * rtw[e * H_ + k];
    lg[e] = s;
    mx = fmaxf(mx, s);
  }
  float sum = 0.0f;
#pragma unroll
  for (int e = 0; e < E_; ++e) { lg[e] = __expf(lg[e] - mx); sum += lg[e]; }
  int i0 = 0;
#pragma unroll
  for (int e = 1; e < E_; ++e) if (lg[e] > lg[i0]) i0 = e;
  int i1 = -1;
#pragma unroll
  for (int e = 0; e < E_; ++e) {
    if (e == i0) continue;
    if (i1 < 0 || lg[e] > lg[i1]) i1 = e;
  }
  const float p0 = lg[i0] / sum, p1 = lg[i1] / sum;
  const float wsum = p0 + p1;
#pragma unroll
  for (int e = 0; e < E_; ++e) gates[t * E_ + e] = 0.0f;
  gates[t * E_ + i0] = p0 / wsum;
  gates[t * E_ + i1] = p1 / wsum;
#pragma unroll
  for (int e = 0; e < E_; ++e)
    atomicAdd(&usage[e], (lg[e] / sum) * (1.0f / B_));
  atomicAdd(&counts[i0], 1.0f);
  atomicAdd(&counts[i1], 1.0f);
  __syncthreads();
  if (t == 0) {
    float aux = 0.0f;
#pragma unroll
    for (int e = 0; e < E_; ++e) aux += usage[e] * (counts[e] / (B_ * 2.0f));
    atomicAdd(auxslot, (float)E_ * aux);
  }
}

// x_moe[b][d] = sum_e gates[b][e] * eo[(e*B + b)*H + d]
__global__ void moe_combine(const float* __restrict__ gates,
                            const float* __restrict__ eo,
                            float* __restrict__ xmoe,
                            bf16* __restrict__ xmoeb) {
  const int b = blockIdx.x, d = threadIdx.x;
  float s = 0.0f;
#pragma unroll
  for (int e = 0; e < E_; ++e)
    s += gates[b * E_ + e] * eo[((long)e * B_ + b) * H_ + d];
  xmoe[b * H_ + d] = s;
  xmoeb[b * H_ + d] = (bf16)s;
}

// ---------------------------------------------------------------------------
// Host-side launch
// ---------------------------------------------------------------------------
static inline long cdiv(long a, long b) { return (a + b - 1) / b; }

extern "C" void kernel_launch(void* const* d_in, const int* in_sizes, int n_in,
                              void* d_out, int out_size, void* d_ws,
                              size_t ws_size, hipStream_t stream) {
  (void)in_sizes; (void)n_in; (void)out_size; (void)ws_size;

  // ---- input mapping (setup_inputs dict order, nested insertion order) ----
  const float* x     = (const float*)d_in[0];
  const float* in_w  = (const float*)d_in[1];
  const float* in_b  = (const float*)d_in[2];
  const float* cw1[3] = {(const float*)d_in[3], (const float*)d_in[7],  (const float*)d_in[11]};
  const float* cb1[3] = {(const float*)d_in[4], (const float*)d_in[8],  (const float*)d_in[12]};
  const float* cw2[3] = {(const float*)d_in[5], (const float*)d_in[9],  (const float*)d_in[13]};
  const float* cb2[3] = {(const float*)d_in[6], (const float*)d_in[10], (const float*)d_in[14]};
  const float* fus_w  = (const float*)d_in[15];
  const float* fus_b  = (const float*)d_in[16];
  const float* fus_g  = (const float*)d_in[17];
  const float* fus_be = (const float*)d_in[18];
  const float* qkv_w  = (const float*)d_in[19];
  const float* qkv_b  = (const float*)d_in[20];
  const float* o_w    = (const float*)d_in[21];
  const float* o_b    = (const float*)d_in[22];
  const float* bln1g[NB_], *bln1b[NB_], *brtw[NB_], *bew1[NB_], *beb1[NB_],
             * bew2[NB_], *beb2[NB_], *bfw1[NB_], *bfb1[NB_], *bfw2[NB_],
             * bfb2[NB_], *bln2g[NB_], *bln2b[NB_];
  for (int nb = 0; nb < NB_; ++nb) {
    const int base = 23 + nb * 13;
    bln1g[nb] = (const float*)d_in[base + 0];
    bln1b[nb] = (const float*)d_in[base + 1];
    brtw[nb]  = (const float*)d_in[base + 2];
    bew1[nb]  = (const float*)d_in[base + 3];
    beb1[nb]  = (const float*)d_in[base + 4];
    bew2[nb]  = (const float*)d_in[base + 5];
    beb2[nb]  = (const float*)d_in[base + 6];
    bfw1[nb]  = (const float*)d_in[base + 7];
    bfb1[nb]  = (const float*)d_in[base + 8];
    bfw2[nb]  = (const float*)d_in[base + 9];
    bfb2[nb]  = (const float*)d_in[base + 10];
    bln2g[nb] = (const float*)d_in[base + 11];
    bln2b[nb] = (const float*)d_in[base + 12];
  }
  const float* d_w1 = (const float*)d_in[49];
  const float* d_b1 = (const float*)d_in[50];
  const float* d_w2 = (const float*)d_in[51];
  const float* d_b2 = (const float*)d_in[52];

  float* out     = (float*)d_out;                    // (B, PRED) then aux
  float* auxslot = out + (long)B_ * PRED_;           // total_aux scalar

  // ---- workspace bump allocator ----
  char*  base = (char*)d_ws;
  size_t off  = 0;
  auto alloc = [&](size_t bytes) -> void* {
    void* p = base + off;
    off += (bytes + 255) & ~(size_t)255;
    return p;
  };
  const int k1s[3] = {3, 4, 8};

  bf16* xbf   = (bf16*)alloc((size_t)BS_ * D_ * 2);
  bf16* w_in  = (bf16*)alloc((size_t)D_ * H_ * 2);          // Bt [256][32]
  bf16* w_c1[3]; bf16* w_c2[3];
  for (int s = 0; s < 3; ++s) {
    w_c1[s] = (bf16*)alloc((size_t)H_ * H_ * k1s[s] * 2);   // Bt [256][k1*256]
    w_c2[s] = (bf16*)alloc((size_t)H_ * H_ * 3 * 2);        // Bt [256][768]
  }
  bf16* w_fus  = (bf16*)alloc((size_t)H3_ * H_ * 2);        // Bt [256][768]
  bf16* w_qkv  = (bf16*)alloc((size_t)H_ * H3_ * 2);        // Bt [768][256] (native)
  bf16* w_o    = (bf16*)alloc((size_t)H_ * H_ * 2);         // Bt [256][256] (native)
  bf16* w_e1[NB_]; bf16* w_e2[NB_]; bf16* w_f1[NB_]; bf16* w_f2[NB_];
  for (int nb = 0; nb < NB_; ++nb) {
    w_e1[nb] = (bf16*)alloc((size_t)E_ * H_ * 2 * H_ * 2);  // Bt [e][512][256]
    w_e2[nb] = (bf16*)alloc((size_t)E_ * 2 * H_ * H_ * 2);  // Bt [e][256][512]
    w_f1[nb] = (bf16*)alloc((size_t)H_ * 4 * H_ * 2);       // Bt [1024][256]
    w_f2[nb] = (bf16*)alloc((size_t)4 * H_ * H_ * 2);       // Bt [256][1024]
  }
  bf16* w_d1 = (bf16*)alloc((size_t)H_ * 2 * H_ * 2);       // Bt [512][256]
  bf16* w_d2 = (bf16*)alloc((size_t)2 * H_ * PRED_ * 2);    // Bt [96][512]

  bf16* xp     = (bf16*)alloc((size_t)BS_ * H_ * 2);     // proj / later q-head
  bf16* tmp    = (bf16*)alloc((size_t)BS_ * H_ * 2);     // conv mid / pre-LN fusion / k-head
  bf16* fcat   = (bf16*)alloc((size_t)BS_ * H3_ * 2);    // concat feats / qkv_lin / ao
  bf16* fusedn = (bf16*)alloc((size_t)BS_ * H_ * 2);     // post-LN fused
  bf16* vhbuf  = (bf16*)alloc((size_t)BS_ * H_ * 2);     // v-head
  float* henc  = (float*)alloc((size_t)B_ * H_ * 4);
  float* xnf   = (float*)alloc((size_t)B_ * H_ * 4);
  bf16*  xnb   = (bf16*)alloc((size_t)B_ * H_ * 2);
  float* gates = (float*)alloc((size_t)B_ * E_ * 4);
  bf16*  hexp  = (bf16*)alloc((size_t)E_ * B_ * 2 * H_ * 2);
  float* eo    = (float*)alloc((size_t)E_ * B_ * H_ * 4);
  float* xmoe  = (float*)alloc((size_t)B_ * H_ * 4);
  bf16*  xmoeb = (bf16*)alloc((size_t)B_ * H_ * 2);
  bf16*  hffn  = (bf16*)alloc((size_t)B_ * 4 * H_ * 2);
  float* ffnout= (float*)alloc((size_t)B_ * H_ * 4);
  float* hbuf1 = (float*)alloc((size_t)B_ * H_ * 4);
  float* hbuf2 = (float*)alloc((size_t)B_ * H_ * 4);
  bf16*  hbf   = (bf16*)alloc((size_t)B_ * H_ * 2);
  bf16*  hdec  = (bf16*)alloc((size_t)B_ * 2 * H_ * 2);

  bf16* qh = xp;      // reuse (xp dead after convs)
  bf16* kh = tmp;     // reuse (tmp dead after encoder LN)
  bf16* ao = fcat;    // reuse (fcat dead after qkv reorder)

  const dim3 blk128(128), blk256(256);

  // ---- weight prep (all B matrices to transposed Bt [N][K] bf16) ----
  cvt_f32_bf16<<<cdiv((long)BS_ * D_, 256), blk256, 0, stream>>>(x, xbf, (long)BS_ * D_);
  transpose_bt_bf16<<<cdiv((long)D_ * H_, 256), blk256, 0, stream>>>(in_w, w_in, D_, H_, (long)D_ * H_);
  for (int s = 0; s < 3; ++s) {
    conv_repack_bt_bf16<<<cdiv((long)H_ * H_ * k1s[s], 256), blk256, 0, stream>>>(cw1[s], w_c1[s], k1s[s]);
    conv_repack_bt_bf16<<<cdiv((long)H_ * H_ * 3, 256), blk256, 0, stream>>>(cw2[s], w_c2[s], 3);
  }
  transpose_bt_bf16<<<cdiv((long)H3_ * H_, 256), blk256, 0, stream>>>(fus_w, w_fus, H3_, H_, (long)H3_ * H_);
  cvt_f32_bf16<<<cdiv((long)H3_ * H_, 256), blk256, 0, stream>>>(qkv_w, w_qkv, (long)H3_ * H_);   // already [N][K]
  cvt_f32_bf16<<<cdiv((long)H_ * H_, 256), blk256, 0, stream>>>(o_w, w_o, (long)H_ * H_);         // already [N][K]
  for (int nb = 0; nb < NB_; ++nb) {
    transpose_bt_bf16<<<cdiv((long)E_ * H_ * 2 * H_, 256), blk256, 0, stream>>>(bew1[nb], w_e1[nb], H_, 2 * H_, (long)E_ * H_ * 2 * H_);
    transpose_bt_bf16<<<cdiv((long)E_ * 2 * H_ * H_, 256), blk256, 0, stream>>>(bew2[nb], w_e2[nb], 2 * H_, H_, (long)E_ * 2 * H_ * H_);
    transpose_bt_bf16<<<cdiv((long)H_ * 4 * H_, 256), blk256, 0, stream>>>(bfw1[nb], w_f1[nb], H_, 4 * H_, (long)H_ * 4 * H_);
    transpose_bt_bf16<<<cdiv((long)4 * H_ * H_, 256), blk256, 0, stream>>>(bfw2[nb], w_f2[nb], 4 * H_, H_, (long)4 * H_ * H_);
  }
  transpose_bt_bf16<<<cdiv((long)H_ * 2 * H_, 256), blk256, 0, stream>>>(d_w1, w_d1, H_, 2 * H_, (long)H_ * 2 * H_);
  transpose_bt_bf16<<<cdiv((long)2 * H_ * PRED_, 256), blk256, 0, stream>>>(d_w2, w_d2, 2 * H_, PRED_, (long)2 * H_ * PRED_);

  // ---- encoder ----
  // input projection: (BS,32) @ (32,256) -> xp
  gemm_wmma_bf16<0, 1, true, false><<<dim3(H_ / 64, BS_ / 64, 1), blk128, 0, stream>>>(
      xbf, D_, w_in, D_, in_b, xp, H_, BS_, H_, D_, 0, nullptr, nullptr, 0, 0, 0, 0);

  // multi-scale convs -> fcat (BS, 768)
  for (int s = 0; s < 3; ++s) {
    const int k1 = k1s[s], pad = k1 / 2;
    gemm_wmma_bf16<1, 1, true, true><<<dim3(H_ / 64, BS_ / 64, 1), blk128, 0, stream>>>(
        xp, H_, w_c1[s], k1 * H_, cb1[s], tmp, H_, BS_, H_, k1 * H_, pad,
        nullptr, nullptr, 0, 0, 0, 0);
    gemm_wmma_bf16<1, 1, true, false><<<dim3(H_ / 64, BS_ / 64, 1), blk128, 0, stream>>>(
        tmp, H_, w_c2[s], 3 * H_, cb2[s], fcat + (long)s * H_, H3_, BS_, H_, 3 * H_, 1,
        nullptr, nullptr, 0, 0, 0, 0);
  }

  // fusion: gelu(fcat @ fus_w + b) -> tmp, then LN -> fusedn
  gemm_wmma_bf16<0, 1, true, true><<<dim3(H_ / 64, BS_ / 64, 1), blk128, 0, stream>>>(
      fcat, H3_, w_fus, H3_, fus_b, tmp, H_, BS_, H_, H3_, 0, nullptr, nullptr, 0, 0, 0, 0);
  ln_bf16_kernel<<<BS_, blk256, 0, stream>>>(tmp, fus_g, fus_be, fusedn);

  // qkv: fusedn @ qkv_w^T + b -> fcat (reused as qkv_lin), then head scatter
  gemm_wmma_bf16<0, 1, true, false><<<dim3(H3_ / 64, BS_ / 64, 1), blk128, 0, stream>>>(
      fusedn, H_, w_qkv, H_, qkv_b, fcat, H3_, BS_, H3_, H_, 0, nullptr, nullptr, 0, 0, 0, 0);
  qkv_reorder<<<cdiv((long)BS_ * H3_, 256), blk256, 0, stream>>>(fcat, qh, kh, vhbuf);

  // attention -> ao (reuses fcat)
  attention_kernel<<<dim3(S_ / 16, B_ * NH_), blk128, 0, stream>>>(qh, kh, vhbuf, ao);

  // o-projection + residual + mean over S -> henc
  zero_f32<<<cdiv((long)B_ * H_, 256), blk256, 0, stream>>>(henc, (long)B_ * H_);
  zero_f32<<<1, 1, 0, stream>>>(auxslot, 1);
  gemm_wmma_bf16<0, 2, true, false><<<dim3(H_ / 64, BS_ / 64, 1), blk128, 0, stream>>>(
      ao, H_, w_o, H_, o_b, nullptr, H_, BS_, H_, H_, 0, fusedn, henc, 0, 0, 0, 0);

  // ---- MoE transformer blocks (256 x 256 state) ----
  float* hcur = henc;
  float* hbufs[2] = {hbuf1, hbuf2};
  for (int nb = 0; nb < NB_; ++nb) {
    ln1_kernel<<<B_, blk256, 0, stream>>>(hcur, bln1g[nb], bln1b[nb], xnf, xnb);
    moe_gate_kernel<<<1, blk256, 0, stream>>>(xnf, brtw[nb], gates, auxslot);
    // experts: h = gelu(xn @ e_w1[e] + e_b1[e])  (batched over e)
    gemm_wmma_bf16<0, 1, true, true><<<dim3(2 * H_ / 64, B_ / 64, E_), blk128, 0, stream>>>(
        xnb, H_, w_e1[nb], H_, beb1[nb], hexp, 2 * H_, B_, 2 * H_, H_, 0,
        nullptr, nullptr, 0, (long)H_ * 2 * H_, (long)B_ * 2 * H_, 2 * H_);
    // eo = h @ e_w2[e] + e_b2[e]
    gemm_wmma_bf16<0, 0, true, false><<<dim3(H_ / 64, B_ / 64, E_), blk128, 0, stream>>>(
        hexp, 2 * H_, w_e2[nb], 2 * H_, beb2[nb], eo, H_, B_, H_, 2 * H_, 0,
        nullptr, nullptr, (long)B_ * 2 * H_, (long)2 * H_ * H_, (long)B_ * H_, H_);
    moe_combine<<<B_, blk256, 0, stream>>>(gates, eo, xmoe, xmoeb);
    // FFN: gelu(xmoe @ f_w1 + b) @ f_w2 + b
    gemm_wmma_bf16<0, 1, true, true><<<dim3(4 * H_ / 64, B_ / 64, 1), blk128, 0, stream>>>(
        xmoeb, H_, w_f1[nb], H_, bfb1[nb], hffn, 4 * H_, B_, 4 * H_, H_, 0,
        nullptr, nullptr, 0, 0, 0, 0);
    gemm_wmma_bf16<0, 0, true, false><<<dim3(H_ / 64, B_ / 64, 1), blk128, 0, stream>>>(
        hffn, 4 * H_, w_f2[nb], 4 * H_, bfb2[nb], ffnout, H_, B_, H_, 4 * H_, 0,
        nullptr, nullptr, 0, 0, 0, 0);
    ln2add_kernel<<<B_, blk256, 0, stream>>>(xmoe, ffnout, bln2g[nb], bln2b[nb], hcur, hbufs[nb]);
    hcur = hbufs[nb];
  }

  // ---- decoder head: gelu(h @ d_w1 + b) @ d_w2 + b -> out ----
  cvt_f32_bf16<<<cdiv((long)B_ * H_, 256), blk256, 0, stream>>>(hcur, hbf, (long)B_ * H_);
  gemm_wmma_bf16<0, 1, true, true><<<dim3(2 * H_ / 64, B_ / 64, 1), blk128, 0, stream>>>(
      hbf, H_, w_d1, H_, d_b1, hdec, 2 * H_, B_, 2 * H_, H_, 0,
      nullptr, nullptr, 0, 0, 0, 0);
  gemm_wmma_bf16<0, 0, true, false><<<dim3((PRED_ + 63) / 64, B_ / 64, 1), blk128, 0, stream>>>(
      hdec, 2 * H_, w_d2, 2 * H_, d_b2, out, PRED_, B_, PRED_, 2 * H_, 0,
      nullptr, nullptr, 0, 0, 0, 0);
}
